// Seq2Seq_32641751450124
// MI455X (gfx1250) — matrix operand order repaired and checked by reading, not compiled, not run on hardware
//
#include <hip/hip_runtime.h>
#include <math.h>

// Seq2Seq LSTM: H=64, F=256, S=8192, T=512
#define HH 64
#define FF 256
#define SS 8192
#define TT 512
#define GG 256   // 4*H gates

typedef __attribute__((ext_vector_type(16))) _Float16 v16h;
typedef __attribute__((ext_vector_type(8)))  float    v8f;

// ---------------------------------------------------------------------------
// Kernel 1: gx[s, n] = sum_k x[s,k] * W_ih_e[n,k] + (b_ih_e[n] + b_hh_e[n])
// One wave per 16x16 output tile; fully unrolled K-loop of 8x
// v_wmma_f32_16x16x32_f16. f32 -> f16 conversion in-register at fragment load.
// ---------------------------------------------------------------------------
__global__ __launch_bounds__(256) void gx_wmma_kernel(
    const float* __restrict__ x,        // (S, F) row-major
    const float* __restrict__ W,        // (4H, F) row-major  (== B^T, contiguous reads)
    const float* __restrict__ b_ih,
    const float* __restrict__ b_hh,
    float* __restrict__ gx)             // (S, 4H) row-major
{
    const int lane  = threadIdx.x & 31;
    const int wave  = threadIdx.x >> 5;
    const int tile  = blockIdx.x * 8 + wave;   // 8192 tiles total
    const int sTile = tile >> 4;               // 512 row tiles over S
    const int nTile = tile & 15;               // 16 col tiles over 4H
    const int s0 = sTile * 16;
    const int n0 = nTile * 16;
    const int lo = lane & 15;
    const int hs = lane >> 4;

    // C/D layout: VGPR r, lane L -> (M = r + 8*hs, N = L&15). Bias constant in M.
    const float bias = b_ih[n0 + lo] + b_hh[n0 + lo];
    v8f acc;
#pragma unroll
    for (int r = 0; r < 8; ++r) acc[r] = bias;

    const float* xrow = x + (size_t)(s0 + lo) * FF;   // A: row M = lane&15
    const float* wrow = W + (size_t)(n0 + lo) * FF;   // B: col N = lane&15

#pragma unroll
    for (int k0 = 0; k0 < FF; k0 += 32) {
        v16h a, b;
        // A 16x32 f16: elems 0..15 = K(k0+hs*16+j), contiguous
#pragma unroll
        for (int j = 0; j < 8; ++j) {
            a[j]     = (_Float16)xrow[k0 + hs * 16 + j];
            a[8 + j] = (_Float16)xrow[k0 + hs * 16 + 8 + j];
        }
        // B 32x16 f16: lane = col N, 16 contiguous K at k0 + hs*16
#pragma unroll
        for (int j = 0; j < 16; ++j) {
            b[j] = (_Float16)wrow[k0 + hs * 16 + j];
        }
        acc = __builtin_amdgcn_wmma_f32_16x16x32_f16(
            /*neg_a=*/false, a, /*neg_b=*/false, b,
            /*c_mod=*/(short)0, acc, /*reuse_a=*/false, /*reuse_b=*/false);
    }

#pragma unroll
    for (int r = 0; r < 8; ++r)
        gx[(size_t)(s0 + r + 8 * hs) * GG + n0 + lo] = acc[r];
}

// ---------------------------------------------------------------------------
// Fast activations: keep the sequential scan's critical path short.
// sigmoid(x) = rcp(1 + exp(-x));  tanh(x) = 1 - 2*rcp(exp(2x) + 1)
// Both lower to v_exp_f32 + v_rcp_f32 (TRANS units) instead of the long
// __ocml polynomial paths.
// ---------------------------------------------------------------------------
__device__ __forceinline__ float sigf(float v) {
    return __builtin_amdgcn_rcpf(1.0f + __expf(-v));
}
__device__ __forceinline__ float tanh_fast(float v) {
    return 1.0f - 2.0f * __builtin_amdgcn_rcpf(__expf(2.0f * v) + 1.0f);
}

// ---------------------------------------------------------------------------
// Kernel 2: sequential encoder (S steps) + decoder (T steps).
// Single 256-thread workgroup (8 waves on one WGP). Thread n owns gate row n:
// its 64-float recurrent weight row lives in registers (fully unrolled FMA
// chain); h / c / gates live in LDS. gx loads are software-pipelined one step
// ahead so L2 latency overlaps the cell + barrier phase.
// ---------------------------------------------------------------------------
__global__ __launch_bounds__(256) void lstm_scan_kernel(
    const float* __restrict__ gx,       // (S, 4H)
    const float* __restrict__ prev_y,   // scalar
    const float* __restrict__ W_hh_e,   // (4H, H)
    const float* __restrict__ W_ih_d,   // (4H, 1)
    const float* __restrict__ W_hh_d,   // (4H, H)
    const float* __restrict__ b_ih_d,   // (4H,)
    const float* __restrict__ b_hh_d,   // (4H,)
    const float* __restrict__ W_out,    // (1, H)
    const float* __restrict__ b_out,    // (1,)
    float* __restrict__ out)            // (T,)
{
    __shared__ float h[HH];
    __shared__ float c[HH];
    __shared__ float gates[GG];
    __shared__ float wout[HH];
    __shared__ float ysh;

    const int tid = threadIdx.x;

    // Encoder recurrent weight row -> registers
    float w[HH];
#pragma unroll
    for (int j = 0; j < HH; ++j) w[j] = W_hh_e[tid * HH + j];

    if (tid < HH) { h[tid] = 0.0f; c[tid] = 0.0f; }
    __syncthreads();

    // ---- encoder: 8192 sequential steps, gx pipelined one step ahead ----
    float gcur = gx[tid];
    for (int t = 0; t < SS; ++t) {
        float acc = 0.0f;
#pragma unroll
        for (int j = 0; j < HH; ++j) acc = fmaf(w[j], h[j], acc);
        gates[tid] = gcur + acc;

        // prefetch next step's gx while the cell + barriers run
        float gnext = 0.0f;
        if (t + 1 < SS) gnext = gx[(size_t)(t + 1) * GG + tid];
        __syncthreads();

        if (tid < HH) {
            const float ig = sigf(gates[tid]);
            const float fg = sigf(gates[HH + tid]);
            const float gg = tanh_fast(gates[2 * HH + tid]);
            const float og = sigf(gates[3 * HH + tid]);
            const float cn = fg * c[tid] + ig * gg;
            c[tid] = cn;
            h[tid] = og * tanh_fast(cn);
        }
        __syncthreads();
        gcur = gnext;
    }

    // ---- decoder setup ----
#pragma unroll
    for (int j = 0; j < HH; ++j) w[j] = W_hh_d[tid * HH + j];
    const float wi = W_ih_d[tid];
    const float bd = b_ih_d[tid] + b_hh_d[tid];
    if (tid < HH) wout[tid] = W_out[tid];
    const float bo = b_out[0];
    if (tid == 0) ysh = prev_y[0];
    __syncthreads();

    // ---- decoder: 512 sequential steps ----
    // Output y[t-1] (a 64-FMA serial reduction on thread 0) is computed
    // concurrently with all threads' gate dot: both depend only on the
    // post-cell h of the previous step.
    for (int t = 0; t < TT; ++t) {
        if (tid == 0 && t > 0) {
            float a = bo;
#pragma unroll
            for (int j = 0; j < HH; ++j) a = fmaf(h[j], wout[j], a);
            const float yo = sigf(a) * 2.0f;
            out[t - 1] = yo;
            ysh = yo;
        }
        float acc = bd;
#pragma unroll
        for (int j = 0; j < HH; ++j) acc = fmaf(w[j], h[j], acc);
        __syncthreads();                      // ysh published

        gates[tid] = fmaf(wi, ysh, acc);
        __syncthreads();                      // gates ready

        if (tid < HH) {
            const float ig = sigf(gates[tid]);
            const float fg = sigf(gates[HH + tid]);
            const float gg = tanh_fast(gates[2 * HH + tid]);
            const float og = sigf(gates[3 * HH + tid]);
            const float cn = fg * c[tid] + ig * gg;
            c[tid] = cn;
            h[tid] = og * tanh_fast(cn);
        }
        __syncthreads();                      // h ready for next step
    }

    // final output (step T-1)
    if (tid == 0) {
        float a = bo;
#pragma unroll
        for (int j = 0; j < HH; ++j) a = fmaf(h[j], wout[j], a);
        out[TT - 1] = sigf(a) * 2.0f;
    }
}

// ---------------------------------------------------------------------------
extern "C" void kernel_launch(void* const* d_in, const int* in_sizes, int n_in,
                              void* d_out, int out_size, void* d_ws, size_t ws_size,
                              hipStream_t stream) {
    (void)in_sizes; (void)n_in; (void)out_size; (void)ws_size;

    const float* x       = (const float*)d_in[0];
    const float* prev_y  = (const float*)d_in[1];
    /* d_in[2] target_y: only defines T (=512, compile-time) */
    const float* W_ih_e  = (const float*)d_in[3];
    const float* W_hh_e  = (const float*)d_in[4];
    const float* b_ih_e  = (const float*)d_in[5];
    const float* b_hh_e  = (const float*)d_in[6];
    const float* W_ih_d  = (const float*)d_in[7];
    const float* W_hh_d  = (const float*)d_in[8];
    const float* b_ih_d  = (const float*)d_in[9];
    const float* b_hh_d  = (const float*)d_in[10];
    const float* W_out   = (const float*)d_in[11];
    const float* b_out   = (const float*)d_in[12];

    float* gx = (float*)d_ws;   // S * 4H floats = 8 MB scratch

    // 8192 16x16 tiles, 8 waves (= 8 tiles) per 256-thread block -> 1024 blocks
    gx_wmma_kernel<<<1024, 256, 0, stream>>>(x, W_ih_e, b_ih_e, b_hh_e, gx);

    // single persistent workgroup runs the sequential encoder + decoder
    lstm_scan_kernel<<<1, 256, 0, stream>>>(gx, prev_y, W_hh_e, W_ih_d, W_hh_d,
                                            b_ih_d, b_hh_d, W_out, b_out,
                                            (float*)d_out);
}
